// PRFSigmoid_23871428231531
// MI455X (gfx1250) — matrix operand order repaired and checked by reading, not compiled, not run on hardware
//

#include <hip/hip_runtime.h>
#include <hip/hip_bf16.h>

// ---- problem constants (match reference) ----
#define DD   1024      // hidden size
#define MM   256       // random features
#define NH   16384     // hidden states
#define KW   8192      // weight rows

typedef __attribute__((ext_vector_type(2)))  float    v2f;
typedef __attribute__((ext_vector_type(4)))  float    f4;
typedef __attribute__((ext_vector_type(8)))  float    v8f;
typedef __attribute__((ext_vector_type(16))) _Float16 v16h;

union HF { v16h h; f4 f[2]; };

// exp(+80), exp(-80): clamp bounds equivalent to clip(log_rf+lc, -80, 80)
#define EXP_P80 5.5406223843935100e+34f
#define EXP_M80 1.8048513878454153e-35f

// =====================================================================
// Phase 1: for each row x of X (nrows x 1024):
//   proj[m]  = x . omega[m]          (m = 0..255)   -- f32 WMMA 16x16x4
//   n2       = ||x||^2
//   e[m]     = proj[m] - n2/2 ;  mx = max_m e[m]
//   phi[m]   = (exp(e[m]-mx) + 1e-6) / 16     -> f16
//   corr     = n2/2 + mx
// One block = 16 rows, 256 threads (8 wave32). Each wave owns two
// 16-wide m-tiles (wave w -> m in [32w, 32w+32)).
// =====================================================================
__global__ __launch_bounds__(256) void prf_phase1(
    const float* __restrict__ X, const float* __restrict__ Om,
    _Float16* __restrict__ phi, float* __restrict__ corr)
{
    __shared__ float s_part[256];
    __shared__ float s_norm2[16];
    __shared__ float s_rmax[16];
    __shared__ float s_expo[16 * 256];

    const int t     = threadIdx.x;
    const int rbase = blockIdx.x * 16;
    const int lane  = t & 31;
    const int wave  = t >> 5;
    const int row16 = t >> 4;   // 0..15
    const int seg   = t & 15;   // 0..15

    // ---- row norms: 16 threads per row, 64 elements each ----
    {
        const float* p = X + (size_t)(rbase + row16) * DD + seg * 64;
        float s = 0.f;
        #pragma unroll
        for (int i = 0; i < 16; ++i) {
            f4 v = *(const f4*)(p + i * 4);
            s += v.x * v.x + v.y * v.y + v.z * v.z + v.w * v.w;
        }
        s_part[t] = s;
    }
    __syncthreads();
    if (t < 16) {
        float s = 0.f;
        #pragma unroll
        for (int i = 0; i < 16; ++i) s += s_part[t * 16 + i];
        s_norm2[t] = s;
    }
    __syncthreads();

#if __has_builtin(__builtin_amdgcn_wmma_f32_16x16x4_f32)
    // ---- projection with V_WMMA_F32_16X16X4_F32 ----
    // A 16x4 f32: lanes 0-15 -> K={0,1}, lanes 16-31 -> K={2,3} (ISA 7.12.2)
    // B  4x16 f32: VGPR0 = K=0 / K=2, VGPR1 = K=1 / K=3 (C/D-style striping)
    {
        const int hi = lane >> 4;      // 0: K-low pair, 1: K-high pair
        const int lr = lane & 15;
        const float* ap  = X  + (size_t)(rbase + lr) * DD          + hi * 2;
        const float* b0p = Om + (size_t)(wave * 32 + lr) * DD      + hi * 2;
        const float* b1p = Om + (size_t)(wave * 32 + 16 + lr) * DD + hi * 2;
        v8f acc0 = {}; v8f acc1 = {};
        #pragma unroll 4
        for (int d = 0; d < DD; d += 4) {
            v2f a  = *(const v2f*)(ap  + d);
            v2f b0 = *(const v2f*)(b0p + d);
            v2f b1 = *(const v2f*)(b1p + d);
            acc0 = __builtin_amdgcn_wmma_f32_16x16x4_f32(
                false, a, false, b0, (short)0, acc0, false, false);
            acc1 = __builtin_amdgcn_wmma_f32_16x16x4_f32(
                false, a, false, b1, (short)0, acc1, false, false);
        }
        // C/D layout: VGPR r, lanes<16 -> M=r ; lanes>=16 -> M=r+8 ; N=lane%16
        #pragma unroll
        for (int r = 0; r < 8; ++r) {
            const int rr  = r + hi * 8;
            const float c = 0.5f * s_norm2[rr];
            s_expo[rr * 256 + wave * 32 + lr]      = acc0[r] - c;
            s_expo[rr * 256 + wave * 32 + 16 + lr] = acc1[r] - c;
        }
    }
    __syncthreads();
#else
    // VALU fallback: thread t computes 16 projections for (row16, seg)
    {
        const float* xp = X + (size_t)(rbase + row16) * DD;
        const float  c  = 0.5f * s_norm2[row16];
        #pragma unroll
        for (int j = 0; j < 16; ++j) {
            const float* wp = Om + (size_t)(seg * 16 + j) * DD;
            float s = 0.f;
            for (int d = 0; d < DD; d += 4) {
                f4 xv = *(const f4*)(xp + d);
                f4 wv = *(const f4*)(wp + d);
                s += xv.x * wv.x + xv.y * wv.y + xv.z * wv.z + xv.w * wv.w;
            }
            s_expo[row16 * 256 + seg * 16 + j] = s - c;
        }
    }
    __syncthreads();
#endif

    // ---- row max over 256 features ----
    {
        float mx = -3.0e38f;
        #pragma unroll
        for (int j = 0; j < 16; ++j)
            mx = fmaxf(mx, s_expo[row16 * 256 + seg * 16 + j]);
        s_part[t] = mx;
    }
    __syncthreads();
    if (t < 16) {
        float mx = s_part[t * 16];
        #pragma unroll
        for (int i = 1; i < 16; ++i) mx = fmaxf(mx, s_part[t * 16 + i]);
        s_rmax[t] = mx;
        corr[rbase + t] = 0.5f * s_norm2[t] + mx;   // (n2)/2 + max
    }
    __syncthreads();

    // ---- phi -> f16 ----
    {
        const float mx = s_rmax[row16];
        _Float16* op = phi + (size_t)(rbase + row16) * MM + seg * 16;
        #pragma unroll
        for (int j = 0; j < 16; ++j) {
            const float e = s_expo[row16 * 256 + seg * 16 + j];
            op[j] = (_Float16)((__expf(e - mx) + 1e-6f) * 0.0625f); // /sqrt(256)
        }
    }
}

// =====================================================================
// Phase 2: S = phi_h (N x 256, f16) @ phi_w^T (256 x K) with f32 acc,
// log-free fused epilogue (exactly equivalent to the reference's
// clip(log(S+1e-10)+lc) -> exp -> sigmoid chain):
//   lc  = clip(hcorr[n] + wcorr[k], -80, 80)
//   e   = clamp((S + 1e-10) * exp(lc), exp(-80), exp(+80))
//   out = e / (1 + e)
// 128x128 C-tile per block; both operand tiles staged in LDS (K = 256
// fully resident). 8 wave32 in a 4(row) x 2(col) grid, each wave does
// a 32x64 sub-tile = 2x4 accumulators of v_wmma_f32_16x16x32_f16.
// =====================================================================
__global__ __launch_bounds__(256) void prf_phase2(
    const _Float16* __restrict__ phiH, const _Float16* __restrict__ phiW,
    const float* __restrict__ hcorr, const float* __restrict__ wcorr,
    float* __restrict__ out)
{
    __shared__ __align__(16) _Float16 As[128 * 256];   // 64 KB
    __shared__ __align__(16) _Float16 Bs[128 * 256];   // 64 KB
    __shared__ float hcs[128];
    __shared__ float wcs[128];

    const int t       = threadIdx.x;
    const int lane    = t & 31;
    const int wave    = t >> 5;
    const int rowBase = blockIdx.y * 128;
    const int colBase = blockIdx.x * 128;

    // Tiles are contiguous 64 KB regions of phi (full 256-feature rows).
    {
        const f4* srcA = (const f4*)(phiH + (size_t)rowBase * 256);
        const f4* srcB = (const f4*)(phiW + (size_t)colBase * 256);
        f4* dstA = (f4*)As;
        f4* dstB = (f4*)Bs;
        #pragma unroll
        for (int i = 0; i < 16; ++i) dstA[t + 256 * i] = srcA[t + 256 * i];
        #pragma unroll
        for (int i = 0; i < 16; ++i) dstB[t + 256 * i] = srcB[t + 256 * i];
    }
    if (t < 128) {
        hcs[t] = hcorr[rowBase + t];
        wcs[t] = wcorr[colBase + t];
    }
    __syncthreads();

    const int hi   = lane >> 4;        // K-half select (ISA 7.12.2)
    const int lr   = lane & 15;
    const int wRow = (wave & 3) * 32;  // 4 waves down
    const int wCol = (wave >> 2) * 64; // 2 waves across

    v8f acc[2][4];
    {
        v8f z = {};
        #pragma unroll
        for (int i = 0; i < 2; ++i)
            #pragma unroll
            for (int j = 0; j < 4; ++j) acc[i][j] = z;
    }

    #pragma unroll
    for (int kk = 0; kk < 256; kk += 32) {
        HF a[2], b[4];
        // A 16x32 f16: lane lr row M=lr; lanes<16 K={0..7,16..23},
        // lanes>=16 K={8..15,24..31} -> two b128 chunks at +hi*8 and +16.
        #pragma unroll
        for (int i = 0; i < 2; ++i) {
            const _Float16* p = As + (wRow + i * 16 + lr) * 256 + kk + hi * 8;
            a[i].f[0] = *(const f4*)(p);
            a[i].f[1] = *(const f4*)(p + 16);
        }
        // B 32x16 f16: lane lr col N=lr; lanes<16 K={0..7,16..23},
        // lanes>=16 K={8..15,24..31} -> two b128 chunks at +hi*8 and +16.
        #pragma unroll
        for (int j = 0; j < 4; ++j) {
            const _Float16* p = Bs + (wCol + j * 16 + lr) * 256 + kk + hi * 8;
            b[j].f[0] = *(const f4*)(p);
            b[j].f[1] = *(const f4*)(p + 16);
        }
        #pragma unroll
        for (int i = 0; i < 2; ++i)
            #pragma unroll
            for (int j = 0; j < 4; ++j)
                acc[i][j] = __builtin_amdgcn_wmma_f32_16x16x32_f16(
                    false, a[i].h, false, b[j].h, (short)0, acc[i][j],
                    false, false);
    }

    // Log-free fused epilogue: one store per element, never re-read.
    #pragma unroll
    for (int i = 0; i < 2; ++i) {
        #pragma unroll
        for (int j = 0; j < 4; ++j) {
            #pragma unroll
            for (int r = 0; r < 8; ++r) {
                const int rl = wRow + i * 16 + r + hi * 8;
                const int cl = wCol + j * 16 + lr;
                const float lc =
                    fminf(fmaxf(hcs[rl] + wcs[cl], -80.f), 80.f);
                float e = (acc[i][j][r] + 1e-10f) * __expf(lc);
                e = fminf(fmaxf(e, EXP_M80), EXP_P80);
                out[(size_t)(rowBase + rl) * KW + (colBase + cl)] =
                    e / (1.f + e);
            }
        }
    }
}

extern "C" void kernel_launch(void* const* d_in, const int* in_sizes, int n_in,
                              void* d_out, int out_size, void* d_ws, size_t ws_size,
                              hipStream_t stream) {
    const float* H  = (const float*)d_in[0];   // (16384, 1024)
    const float* W  = (const float*)d_in[1];   // (8192, 1024)
    const float* Om = (const float*)d_in[2];   // (256, 1024)
    float* out = (float*)d_out;                // (16384, 8192) f32

    char* ws = (char*)d_ws;
    _Float16* phiH = (_Float16*)ws;                               // 8 MB
    _Float16* phiW = (_Float16*)(ws + (size_t)NH * MM * 2);       // 4 MB
    float*    hc   = (float*)   (ws + (size_t)(NH + KW) * MM * 2);// 64 KB
    float*    wc   = hc + NH;                                     // 32 KB

    prf_phase1<<<NH / 16, 256, 0, stream>>>(H, Om, phiH, hc);
    prf_phase1<<<KW / 16, 256, 0, stream>>>(W, Om, phiW, wc);
    prf_phase2<<<dim3(KW / 128, NH / 128), 256, 0, stream>>>(phiH, phiW, hc, wc, out);
}
